// GIN_mini_10213432229998
// MI455X (gfx1250) — matrix operand order repaired and checked by reading, not compiled, not run on hardware
//
#include <hip/hip_runtime.h>
#include <hip/hip_bf16.h>

typedef __attribute__((ext_vector_type(16))) _Float16 v16h;
typedef __attribute__((ext_vector_type(8)))  _Float16 v8h;
typedef __attribute__((ext_vector_type(2)))  _Float16 h2v;
typedef __attribute__((ext_vector_type(8)))  float    v8f;
typedef int v4i __attribute__((vector_size(16)));

#define FEAT 128
#define TM   64          // rows per GEMM block
#define ROWW 136         // LDS row pitch in halfs (272B -> conflict-free b128 frag loads)
#define MATSZ (FEAT * ROWW)   // halfs per pre-transposed weight matrix
#define BN_EPS 1e-5f

// ---------------------------------------------------------------------------
// async global -> LDS b128 copy (ASYNCcnt path), with plain-copy fallback.
// Builtin signature (from toolchain diagnostic): param0 = v4i* in AS1 (global),
// param1 = v4i* in AS3 (LDS), then imm offset, imm cpol.
// ---------------------------------------------------------------------------
#if __has_builtin(__builtin_amdgcn_global_load_async_to_lds_b128) && \
    __has_builtin(__builtin_amdgcn_s_wait_asynccnt)
#define HAVE_ASYNC_LDS 1
__device__ inline void async_cp16(const void* g, void* l) {
    __builtin_amdgcn_global_load_async_to_lds_b128(
        (__attribute__((address_space(1))) v4i*)g,
        (__attribute__((address_space(3))) v4i*)l, 0, 0);
}
#endif

// ---------------------------------------------------------------------------
// packed f16 atomic add (global_atomic_pk_add_f16); CAS fallback
// ---------------------------------------------------------------------------
__device__ inline void pk_atomic_add_f16(unsigned int* addr, unsigned int val) {
#if __has_builtin(__builtin_amdgcn_flat_atomic_fadd_v2f16)
    (void)__builtin_amdgcn_flat_atomic_fadd_v2f16((h2v*)addr, __builtin_bit_cast(h2v, val));
#else
    unsigned int old = *addr, assumed;
    do {
        assumed = old;
        h2v s = __builtin_bit_cast(h2v, assumed) + __builtin_bit_cast(h2v, val);
        old = atomicCAS(addr, assumed, __builtin_bit_cast(unsigned int, s));
    } while (old != assumed);
#endif
}

// ---------------------------------------------------------------------------
// x (f32) -> xh (f16), vectorized
// ---------------------------------------------------------------------------
__global__ void cvt_kernel(const float* __restrict__ x, _Float16* __restrict__ xh, int n4) {
    int i = blockIdx.x * blockDim.x + threadIdx.x;
    if (i < n4) {
        float4 v = ((const float4*)x)[i];
        h2v a = {(_Float16)v.x, (_Float16)v.y};
        h2v b = {(_Float16)v.z, (_Float16)v.w};
        ((h2v*)xh)[2 * i]     = a;
        ((h2v*)xh)[2 * i + 1] = b;
    }
}

// agg = x (f16, uint4 chunks = 8 halfs)
__global__ void copy_h_kernel(const _Float16* __restrict__ src, _Float16* __restrict__ dst, int n8) {
    int i = blockIdx.x * blockDim.x + threadIdx.x;
    if (i < n8) ((uint4*)dst)[i] = ((const uint4*)src)[i];
}

// ---------------------------------------------------------------------------
// agg[dst[e]] += x[src[e]]  (f16): one wave per edge, 4 halfs per lane,
// two packed-f16 atomics per lane (256B row per edge)
// ---------------------------------------------------------------------------
__global__ void scatter_h_kernel(const _Float16* __restrict__ x,
                                 const int* __restrict__ srcI,
                                 const int* __restrict__ dstI,
                                 _Float16* __restrict__ agg, int E) {
    int gtid = blockIdx.x * blockDim.x + threadIdx.x;
    int e    = gtid >> 5;
    int lane = gtid & 31;
    if (e >= E) return;
    int s = srcI[e];
    int d = dstI[e];
    uint2 v = ((const uint2*)(x + (size_t)s * FEAT))[lane];
    unsigned int* dp = (unsigned int*)(agg + (size_t)d * FEAT) + lane * 2;
    pk_atomic_add_f16(dp + 0, v.x);
    pk_atomic_add_f16(dp + 1, v.y);
}

__global__ void zero_kernel(float* __restrict__ p, int n) {
    int i = blockIdx.x * blockDim.x + threadIdx.x;
    if (i < n) p[i] = 0.0f;
}

// ---------------------------------------------------------------------------
// pre-transpose + convert weights: Wt[mat][n*ROWW + k] = (f16) W[mat][k*128 + n]
// ---------------------------------------------------------------------------
__global__ void wprep_kernel(const float* __restrict__ W, _Float16* __restrict__ Wt, int total) {
    int i = blockIdx.x * blockDim.x + threadIdx.x;
    if (i < total) {
        int mat = i >> 14;
        int r   = i & 16383;
        int n = r >> 7, k = r & 127;
        Wt[(size_t)mat * MATSZ + n * ROWW + k] = (_Float16)W[(size_t)mat * 16384 + k * FEAT + n];
    }
}

// ---------------------------------------------------------------------------
// BN stats -> per-feature affine
// ---------------------------------------------------------------------------
__global__ void bnprep_kernel(const float* __restrict__ stats,
                              const float* __restrict__ gamma,
                              const float* __restrict__ beta,
                              float* __restrict__ scale,
                              float* __restrict__ shift, float invN) {
    int i = threadIdx.x;   // 128 threads
    float mean = stats[i] * invN;
    float var  = fmaxf(stats[FEAT + i] * invN - mean * mean, 0.0f);
    float inv  = rsqrtf(var + BN_EPS);
    float sc   = gamma[i] * inv;
    scale[i] = sc;
    shift[i] = beta[i] - mean * sc;
}

// ---------------------------------------------------------------------------
// C[N,128] = relu( op(A)[N,128] @ W + bias ), A/C f16, f32 accum WMMA.
// W^T and (non-BN) A tiles staged via async global->LDS DMA when available.
// ---------------------------------------------------------------------------
template <bool BN_IN, bool STATS>
__global__ __launch_bounds__(256)
void gemm_kernel(const _Float16* __restrict__ A, const _Float16* __restrict__ Wt,
                 const float* __restrict__ bias,
                 const float* __restrict__ bnScale, const float* __restrict__ bnShift,
                 _Float16* __restrict__ out, float* __restrict__ stats, int N) {
    __shared__ __align__(16) _Float16 sW[FEAT * ROWW];   // W^T f16, pitched
    __shared__ __align__(16) _Float16 sA[TM * ROWW];     // A tile f16, pitched
    __shared__ float sSum[FEAT];
    __shared__ float sSq[FEAT];

    const int tid  = threadIdx.x;
    const int base = blockIdx.x * TM;
    const bool full = (base + TM) <= N;

    // Stage W^T (raw b128 chunks; pre-transposed/converted/pitched).
#ifdef HAVE_ASYNC_LDS
    for (int i = tid; i < MATSZ / 8; i += 256)
        async_cp16(Wt + 8 * i, sW + 8 * i);
#else
    for (int i = tid; i < MATSZ / 8; i += 256)
        ((uint4*)sW)[i] = ((const uint4*)Wt)[i];
#endif

    // Stage A tile.
    if (BN_IN) {
        if (full) {
            for (int i = tid; i < TM * FEAT; i += 256) {
                int m = i >> 7, k = i & 127;
                float v = (float)A[(size_t)(base + m) * FEAT + k];
                sA[m * ROWW + k] = (_Float16)(v * bnScale[k] + bnShift[k]);
            }
        } else {
            for (int i = tid; i < TM * FEAT; i += 256) {
                int m = i >> 7, k = i & 127;
                int row = base + m;
                float v = (row < N) ? (float)A[(size_t)row * FEAT + k] : 0.0f;
                sA[m * ROWW + k] = (_Float16)(v * bnScale[k] + bnShift[k]);
            }
        }
    } else {
        if (full) {
#ifdef HAVE_ASYNC_LDS
            for (int i = tid; i < TM * 16; i += 256) {
                int m = i >> 4, j = i & 15;
                async_cp16(A + (size_t)(base + m) * FEAT + 8 * j, sA + m * ROWW + 8 * j);
            }
#else
            for (int i = tid; i < TM * 16; i += 256) {
                int m = i >> 4, j = i & 15;
                ((uint4*)sA)[m * 17 + j] =
                    ((const uint4*)(A + (size_t)(base + m) * FEAT))[j];
            }
#endif
        } else {
            for (int i = tid; i < TM * 16; i += 256) {
                int m = i >> 4, j = i & 15;
                int row = base + m;
                uint4 z = {0u, 0u, 0u, 0u};
                ((uint4*)sA)[m * 17 + j] =
                    (row < N) ? ((const uint4*)(A + (size_t)row * FEAT))[j] : z;
            }
        }
    }
    if (STATS && tid < FEAT) { sSum[tid] = 0.0f; sSq[tid] = 0.0f; }
#ifdef HAVE_ASYNC_LDS
    __builtin_amdgcn_s_wait_asynccnt(0);
#endif
    __syncthreads();

    const int wid  = tid >> 5;     // col tile 0..7
    const int lane = tid & 31;
    const int hh   = lane >> 4;
    const int ln   = lane & 15;
    const int n_g  = wid * 16 + ln;
    const float bv = bias[n_g];

    float lsum = 0.0f, lsq = 0.0f;

    for (int rt = 0; rt < TM / 16; ++rt) {
        v8f acc = {};
#pragma unroll
        for (int kk = 0; kk < 4; ++kk) {
            const int kb = kk * 32;
            const v8h a0 = *(const v8h*)&sA[(rt * 16 + ln) * ROWW + kb + hh * 8];
            const v8h a1 = *(const v8h*)&sA[(rt * 16 + ln) * ROWW + kb + 16 + hh * 8];
            const v8h b0 = *(const v8h*)&sW[n_g * ROWW + kb + hh * 8];
            const v8h b1 = *(const v8h*)&sW[n_g * ROWW + kb + 16 + hh * 8];
            v16h av, bw;
#pragma unroll
            for (int e = 0; e < 8; ++e) {
                av[e] = a0[e]; av[e + 8] = a1[e];
                bw[e] = b0[e]; bw[e + 8] = b1[e];
            }
            acc = __builtin_amdgcn_wmma_f32_16x16x32_f16(
                false, av, false, bw, (short)0, acc, false, false);
        }
        // bias + ReLU + f16 store (C layout: elem j -> row 8*hh+j, col ln)
        if (full) {
#pragma unroll
            for (int j = 0; j < 8; ++j) {
                int row = base + rt * 16 + hh * 8 + j;
                float val = fmaxf(acc[j] + bv, 0.0f);
                out[(size_t)row * FEAT + n_g] = (_Float16)val;
                if (STATS) { lsum += val; lsq += val * val; }
            }
        } else {
#pragma unroll
            for (int j = 0; j < 8; ++j) {
                int row = base + rt * 16 + hh * 8 + j;
                if (row < N) {
                    float val = fmaxf(acc[j] + bv, 0.0f);
                    out[(size_t)row * FEAT + n_g] = (_Float16)val;
                    if (STATS) { lsum += val; lsq += val * val; }
                }
            }
        }
    }

    if (STATS) {
        atomicAdd(&sSum[n_g], lsum);   // ds_add_f32
        atomicAdd(&sSq[n_g], lsq);
        __syncthreads();
        if (tid < FEAT) {
            unsafeAtomicAdd(&stats[tid], sSum[tid]);
            unsafeAtomicAdd(&stats[FEAT + tid], sSq[tid]);
        }
    }
}

// ---------------------------------------------------------------------------
// out[row] = dot(h[row,:], w) + b  — one wave per row, h in f16
// ---------------------------------------------------------------------------
__global__ void fc2_kernel(const _Float16* __restrict__ h, const float* __restrict__ w,
                           const float* __restrict__ b, float* __restrict__ out, int N) {
    __shared__ float sw[FEAT];
    int tid = threadIdx.x;
    if (tid < FEAT) sw[tid] = w[tid];
    __syncthreads();
    int lane = tid & 31, wid = tid >> 5;
    int row = blockIdx.x * 8 + wid;
    if (row >= N) return;
    const h2v* hr = (const h2v*)(h + (size_t)row * FEAT);
    h2v p0 = hr[2 * lane], p1 = hr[2 * lane + 1];
    float acc = (float)p0[0] * sw[4 * lane]     + (float)p0[1] * sw[4 * lane + 1] +
                (float)p1[0] * sw[4 * lane + 2] + (float)p1[1] * sw[4 * lane + 3];
#pragma unroll
    for (int off = 16; off > 0; off >>= 1) acc += __shfl_down(acc, off, 32);
    if (lane == 0) out[row] = acc + b[0];
}

// ---------------------------------------------------------------------------
extern "C" void kernel_launch(void* const* d_in, const int* in_sizes, int n_in,
                              void* d_out, int out_size, void* d_ws, size_t ws_size,
                              hipStream_t stream) {
    const float* x     = (const float*)d_in[0];
    const int*   ei    = (const int*)d_in[1];
    const float* w1    = (const float*)d_in[2];
    const float* b1    = (const float*)d_in[3];
    const float* gamma = (const float*)d_in[4];
    const float* beta  = (const float*)d_in[5];
    const float* w2    = (const float*)d_in[6];
    const float* b2    = (const float*)d_in[7];
    const float* fc1w  = (const float*)d_in[8];
    const float* fc1b  = (const float*)d_in[9];
    const float* fc2w  = (const float*)d_in[10];
    const float* fc2b  = (const float*)d_in[11];

    const int N = in_sizes[0] / FEAT;
    const int E = in_sizes[1] / 2;
    const int* srcI = ei;
    const int* dstI = ei + E;

    const size_t NB = (size_t)N * FEAT;         // halfs per activation buffer
    _Float16* Xh = (_Float16*)d_ws;
    _Float16* A0 = Xh + NB;
    _Float16* A1 = A0 + NB;
    _Float16* A2 = A1 + NB;
    _Float16* Wt = A2 + NB;                     // 9 * MATSZ halfs
    float* stats   = (float*)(Wt + 9 * (size_t)MATSZ);
    float* bnScale = stats + 256;
    float* bnShift = bnScale + FEAT;

    const dim3 blk(256);
    const int gemmGrid    = (N + TM - 1) / TM;
    const int cvtGrid     = (N * 32 + 255) / 256;
    const int copyGrid    = (N * 16 + 255) / 256;
    const int scatterGrid = (int)(((long)E * 32 + 255) / 256);
    const float invN      = 1.0f / (float)N;

    // one-time per call: weights -> f16, transposed + pitched; x -> f16
    wprep_kernel<<<(4 * 16384 + 255) / 256, blk, 0, stream>>>(w1, Wt, 4 * 16384);
    wprep_kernel<<<(4 * 16384 + 255) / 256, blk, 0, stream>>>(w2, Wt + 4 * (size_t)MATSZ, 4 * 16384);
    wprep_kernel<<<(16384 + 255) / 256, blk, 0, stream>>>(fc1w, Wt + 8 * (size_t)MATSZ, 16384);
    cvt_kernel<<<cvtGrid, blk, 0, stream>>>(x, Xh, N * 32);

    const _Float16* xcur = Xh;
    for (int l = 0; l < 4; ++l) {
        _Float16* agg  = (l & 1) ? A1 : A0;
        _Float16* h1   = A2;
        _Float16* outb = agg;   // agg dead after GEMM1 -> reuse for layer output

        copy_h_kernel<<<copyGrid, blk, 0, stream>>>(xcur, agg, N * 16);
        scatter_h_kernel<<<scatterGrid, blk, 0, stream>>>(xcur, srcI, dstI, agg, E);
        zero_kernel<<<1, blk, 0, stream>>>(stats, 256);
        gemm_kernel<false, true><<<gemmGrid, blk, 0, stream>>>(
            agg, Wt + (size_t)l * MATSZ, b1 + l * FEAT,
            nullptr, nullptr, h1, stats, N);
        bnprep_kernel<<<1, FEAT, 0, stream>>>(stats, gamma + l * FEAT, beta + l * FEAT,
                                              bnScale, bnShift, invN);
        gemm_kernel<true, false><<<gemmGrid, blk, 0, stream>>>(
            h1, Wt + (size_t)(4 + l) * MATSZ, b2 + l * FEAT,
            bnScale, bnShift, outb, nullptr, N);
        xcur = outb;
    }

    // fc1: relu(h @ fc1_w + fc1_b)
    _Float16* hfc = (xcur == A1) ? A0 : A1;
    gemm_kernel<false, false><<<gemmGrid, blk, 0, stream>>>(
        xcur, Wt + 8 * (size_t)MATSZ, fc1b, nullptr, nullptr, hfc, nullptr, N);
    // fc2: [N,1] f32
    fc2_kernel<<<(N + 7) / 8, blk, 0, stream>>>(hfc, fc2w, fc2b, (float*)d_out, N);
}
